// HyperAttention_89910845375083
// MI455X (gfx1250) — matrix operand order repaired and checked by reading, not compiled, not run on hardware
//
#include <hip/hip_runtime.h>
#include <hip/hip_bf16.h>
#include <cfloat>

// ---------------------------------------------------------------------------
// HyperAttention fused pipeline for MI455X (gfx1250, wave32, WMMA).
//
// Stage 1: d_att = drug_conv @ W_d^T + b_d          (WMMA f16->f32, stripe)
//          p_att = prot_conv @ W_p^T + b_p
// Stage 2: mean_d[b,i,d] = (1/LP) sum_j relu(d_att[b,i,d]+p_att[b,j,d])
//          mean_p[b,j,d] = (1/LD) sum_i relu(...)   (fused, M never built)
// Stage 3: catt = sigmoid(mean_d @ W_att^T + b_att) (WMMA)
//          patt = sigmoid(mean_p @ W_att^T + b_att)
// Stage 4: out = LayerNorm(max_L(conv * (0.5 + att)))
//
// v2: one wave now computes a full 16x128 output stripe -> A loaded and
// converted to f16 exactly once (8x less A traffic / cvt VALU), 32 WMMA per
// kernel body. Streaming reducers use global_prefetch_b8 to cover L2 latency.
// ---------------------------------------------------------------------------

typedef __attribute__((ext_vector_type(16))) _Float16 v16h;
typedef __attribute__((ext_vector_type(8)))  float    v8f;

#define BATCH 8
#define LD    128
#define LP    1000
#define DIM   128

// out[r, :] = act( X[r,:] @ W^T + bias ),  X: R x 128, W: 128 x 128.
// grid = R/16, block = 32: one wave owns a 16x128 output stripe.
__global__ __launch_bounds__(32)
void linear_wmma_kernel(const float* __restrict__ X,
                        const float* __restrict__ W,
                        const float* __restrict__ bias,
                        float* __restrict__ out,
                        int applySigmoid)
{
    const int lane = threadIdx.x;
    const int lo   = lane & 15;   // A: row M, B/C/D: col N
    const int hi   = lane >> 4;   // half-wave select
    const int row0 = blockIdx.x * 16;

    const float* xrow = X + (size_t)(row0 + lo) * DIM;

    // ---- Load + convert the 16x128 A stripe once (4 K-tiles of 16x32 f16) ----
    v16h a[4];
    #pragma unroll
    for (int kt = 0; kt < 4; ++kt) {
        const int kk = kt * 32;
        #pragma unroll
        for (int v = 0; v < 8; ++v) {
            // A 16x32 f16 layout (ISA 7.12.2): VGPR v holds a K pair;
            // lanes 16-31 offset K by +8 (low half) / +24 (high half).
            const int ka = (v < 4) ? (hi * 8 + 2 * v) : (16 + hi * 8 + 2 * (v - 4));
            a[kt][2 * v]     = (_Float16)xrow[kk + ka];
            a[kt][2 * v + 1] = (_Float16)xrow[kk + ka + 1];
        }
    }

    v8f acc[8];
    #pragma unroll
    for (int ct = 0; ct < 8; ++ct) acc[ct] = (v8f){};

    // ---- 8 column tiles x 4 K-tiles = 32 WMMA ----
    #pragma unroll
    for (int ct = 0; ct < 8; ++ct) {
        const int col0 = ct * 16;
        const float* wrow = W + (size_t)(col0 + lo) * DIM; // B col n=lo <- W row col0+lo
        #pragma unroll
        for (int kt = 0; kt < 4; ++kt) {
            const int kk = kt * 32;
            v16h bm;
            #pragma unroll
            for (int v = 0; v < 8; ++v) {
                // B 32x16 f16 layout: same K order as A (lanes 16-31 offset K by +8 / +24).
                const int kb = (v < 4) ? (hi * 8 + 2 * v) : (16 + hi * 8 + 2 * (v - 4));
                bm[2 * v]     = (_Float16)wrow[kk + kb];
                bm[2 * v + 1] = (_Float16)wrow[kk + kb + 1];
            }
            acc[ct] = __builtin_amdgcn_wmma_f32_16x16x32_f16(
                /*neg_a=*/false, a[kt], /*neg_b=*/false, bm,
                /*c_mod=*/(short)0, acc[ct], /*reuse_a=*/false, /*reuse_b=*/false);
        }
    }

    // ---- Epilogue: bias (+ sigmoid), store stripe ----
    #pragma unroll
    for (int ct = 0; ct < 8; ++ct) {
        const int col0 = ct * 16;
        const float bv = bias[col0 + lo];
        #pragma unroll
        for (int v = 0; v < 8; ++v) {
            // C/D layout: VGPR v -> row M = v + 8*hi, col N = lo.
            const int r = row0 + v + 8 * hi;
            float val = acc[ct][v] + bv;
            if (applySigmoid) val = 1.0f / (1.0f + __expf(-val));
            out[(size_t)r * DIM + (col0 + lo)] = val;
        }
    }
}

// mean over LP axis: 8 stationary d_att rows per block, stream p_att once.
// grid = B*LD/8 = 128, block = 128 (thread t owns dim t).
__global__ __launch_bounds__(128)
void mean_over_protein_kernel(const float* __restrict__ d_att,
                              const float* __restrict__ p_att,
                              float* __restrict__ mean_d)
{
    const int b  = blockIdx.x >> 4;          // 16 row-groups per batch
    const int i0 = (blockIdx.x & 15) * 8;
    const int t  = threadIdx.x;

    float a[8], acc[8];
    #pragma unroll
    for (int r = 0; r < 8; ++r) {
        a[r]   = d_att[(size_t)(b * LD + i0 + r) * DIM + t];
        acc[r] = 0.0f;
    }
    const float* pp = p_att + (size_t)b * LP * DIM + t;
    for (int j = 0; j < LP; ++j) {
        __builtin_prefetch(&pp[(size_t)(j + 16) * DIM], 0, 0); // global_prefetch_b8
        const float p = pp[(size_t)j * DIM];
        #pragma unroll
        for (int r = 0; r < 8; ++r)
            acc[r] += fmaxf(a[r] + p, 0.0f);
    }
    #pragma unroll
    for (int r = 0; r < 8; ++r)
        mean_d[(size_t)(b * LD + i0 + r) * DIM + t] = acc[r] * (1.0f / LP);
}

// mean over LD axis: 8 stationary p_att rows per block, stream d_att once.
// grid = B*LP/8 = 1000, block = 128.
__global__ __launch_bounds__(128)
void mean_over_drug_kernel(const float* __restrict__ d_att,
                           const float* __restrict__ p_att,
                           float* __restrict__ mean_p)
{
    const int b  = blockIdx.x / 125;         // 125 row-groups per batch
    const int j0 = (blockIdx.x % 125) * 8;
    const int t  = threadIdx.x;

    float p[8], acc[8];
    #pragma unroll
    for (int r = 0; r < 8; ++r) {
        p[r]   = p_att[(size_t)(b * LP + j0 + r) * DIM + t];
        acc[r] = 0.0f;
    }
    const float* dd = d_att + (size_t)b * LD * DIM + t;
    for (int i = 0; i < LD; ++i) {
        __builtin_prefetch(&dd[(size_t)(i + 16) * DIM], 0, 0);
        const float aa = dd[(size_t)i * DIM];
        #pragma unroll
        for (int r = 0; r < 8; ++r)
            acc[r] += fmaxf(p[r] + aa, 0.0f);
    }
    #pragma unroll
    for (int r = 0; r < 8; ++r)
        mean_p[(size_t)(b * LP + j0 + r) * DIM + t] = acc[r] * (1.0f / LD);
}

// out[b,:] = LayerNorm_dim( max_i( conv[b,i,:] * (0.5 + att[b,i,:]) ) )
// grid = B, block = 128.
__global__ __launch_bounds__(128)
void gate_max_layernorm_kernel(const float* __restrict__ conv,
                               const float* __restrict__ att,
                               const float* __restrict__ gamma,
                               const float* __restrict__ beta,
                               float* __restrict__ out, int L)
{
    __shared__ float red[DIM];
    const int b = blockIdx.x;
    const int t = threadIdx.x;

    const float* c = conv + (size_t)b * L * DIM + t;
    const float* g = att  + (size_t)b * L * DIM + t;
    float m = -FLT_MAX;
    for (int i = 0; i < L; ++i) {
        const float x  = c[(size_t)i * DIM];
        const float av = g[(size_t)i * DIM];
        m = fmaxf(m, x * (0.5f + av));
    }

    // mean
    red[t] = m;
    __syncthreads();
    for (int s = 64; s > 0; s >>= 1) {
        if (t < s) red[t] += red[t + s];
        __syncthreads();
    }
    const float mu = red[0] * (1.0f / DIM);
    __syncthreads();

    // variance
    const float d = m - mu;
    red[t] = d * d;
    __syncthreads();
    for (int s = 64; s > 0; s >>= 1) {
        if (t < s) red[t] += red[t + s];
        __syncthreads();
    }
    const float var = red[0] * (1.0f / DIM);

    out[(size_t)b * DIM + t] = d * rsqrtf(var + 1e-5f) * gamma[t] + beta[t];
}

extern "C" void kernel_launch(void* const* d_in, const int* in_sizes, int n_in,
                              void* d_out, int out_size, void* d_ws, size_t ws_size,
                              hipStream_t stream)
{
    (void)in_sizes; (void)n_in; (void)out_size; (void)ws_size;

    const float* drug_conv = (const float*)d_in[0];   // (8,128,128)
    const float* prot_conv = (const float*)d_in[1];   // (8,1000,128)
    const float* W_d   = (const float*)d_in[2];
    const float* b_d   = (const float*)d_in[3];
    const float* W_p   = (const float*)d_in[4];
    const float* b_p   = (const float*)d_in[5];
    const float* W_att = (const float*)d_in[6];
    const float* b_att = (const float*)d_in[7];
    const float* gamma1= (const float*)d_in[8];
    const float* beta1 = (const float*)d_in[9];
    const float* gamma2= (const float*)d_in[10];
    const float* beta2 = (const float*)d_in[11];
    float* out = (float*)d_out;

    // Workspace layout (floats); total ~13.9 MB, fully L2-resident (192 MB).
    float* ws     = (float*)d_ws;
    float* d_att  = ws;
    float* p_att  = d_att  + (size_t)BATCH * LD * DIM;
    float* mean_d = p_att  + (size_t)BATCH * LP * DIM;
    float* mean_p = mean_d + (size_t)BATCH * LD * DIM;
    float* catt   = mean_p + (size_t)BATCH * LP * DIM;
    float* patt   = catt   + (size_t)BATCH * LD * DIM;

    const dim3 wblk(32);
    // Stage 1: input projections (WMMA stripes)
    linear_wmma_kernel<<<(BATCH * LD) / 16, wblk, 0, stream>>>(
        drug_conv, W_d, b_d, d_att, 0);
    linear_wmma_kernel<<<(BATCH * LP) / 16, wblk, 0, stream>>>(
        prot_conv, W_p, b_p, p_att, 0);

    // Stage 2: fused dual reduction of relu(d+p).
    mean_over_protein_kernel<<<(BATCH * LD) / 8, 128, 0, stream>>>(d_att, p_att, mean_d);
    mean_over_drug_kernel<<<(BATCH * LP) / 8, 128, 0, stream>>>(d_att, p_att, mean_p);

    // Stage 3: attention projection + sigmoid (WMMA stripes)
    linear_wmma_kernel<<<(BATCH * LD) / 16, wblk, 0, stream>>>(
        mean_d, W_att, b_att, catt, 1);
    linear_wmma_kernel<<<(BATCH * LP) / 16, wblk, 0, stream>>>(
        mean_p, W_att, b_att, patt, 1);

    // Stage 4: gate, max-pool over length, LayerNorm.
    gate_max_layernorm_kernel<<<BATCH, DIM, 0, stream>>>(
        drug_conv, catt, gamma1, beta1, out, LD);
    gate_max_layernorm_kernel<<<BATCH, DIM, 0, stream>>>(
        prot_conv, patt, gamma2, beta2, out + BATCH * DIM, LP);
}
